// ObsEncoder_72198400246036
// MI455X (gfx1250) — matrix operand order repaired and checked by reading, not compiled, not run on hardware
//
#include <hip/hip_runtime.h>
#include <hip/hip_bf16.h>

// ---------------- types ----------------
typedef _Float16 half_t;
typedef __attribute__((ext_vector_type(16))) _Float16 v16h;
typedef __attribute__((ext_vector_type(8)))  _Float16 v8h;
typedef __attribute__((ext_vector_type(8)))  float    v8f;
typedef __attribute__((ext_vector_type(4)))  unsigned uv4;

union F16x16 { uv4 u[2]; v16h h; };

// ---------------- problem constants ----------------
#define OBS_DIM 42
#define HDIM    128
#define BATCH   524288

// f16 weight workspace layout (element offsets), all matrices stored
// TRANSPOSED as N x K (K padded to multiple of 32), row-major in K.
#define WS_SELFW 0          // 128 x 32  (K: 17 -> 32)
#define WS_ENW   4096       // 128 x 32  (K: 5  -> 32)
#define WS_ALW   8192       // 128 x 32  (K: 5  -> 32)
#define WS_EMAT  12288      // 128 x 128
#define WS_AMAT  28672      // 128 x 128
#define WS_ENC1  45056      // 128 x 384
#define WS_ENC2  94208      // 128 x 128
#define WS_TOTAL 110592     // halves (221,184 bytes)

#define WAVES_PER_WG 4
#define ROWS_PER_WAVE 16
#define LDS_PER_WAVE 36864  // bytes
#define LDS_TOTAL (LDS_PER_WAVE * WAVES_PER_WG)

// ---------------- helpers ----------------
__device__ __forceinline__ float act_tanh(float x) {
#if __has_builtin(__builtin_amdgcn_tanhf)
    return __builtin_amdgcn_tanhf(x);            // CDNA5 V_TANH_F32 (TRANS op)
#else
    x = fminf(fmaxf(x, -10.f), 10.f);
    float t = __builtin_amdgcn_exp2f(x * 2.885390081777927f); // exp(2x)
    return 1.f - 2.f / (t + 1.f);
#endif
}

#define WMMA_F16(a, b, c) \
    __builtin_amdgcn_wmma_f32_16x16x32_f16(false, (a), false, (b), (short)0, (c), false, false)

// A fragment: 16x32 f16, A in LDS row-major (M x K), stride in halves.
// lanes 0-15: V0-3 = K 0..7, V4-7 = K 16..23 ; lanes 16-31: +8 on K.
__device__ __forceinline__ v16h frag_a(const half_t* A, int stride, int k0, int lane) {
    const half_t* p = A + (lane & 15) * stride + k0 + ((lane >> 4) << 3);
    F16x16 f;
    f.u[0] = *(const uv4*)(p);
    f.u[1] = *(const uv4*)(p + 16);
    return f.h;
}

// B fragment: 32x16 f16, B stored transposed (N x K) row-major, kstride halves.
// lane n (0-15): K 0..7, 16..23 of column n0+n ; lanes 16-31: +8 on K.
__device__ __forceinline__ v16h frag_b(const half_t* BT, int kstride, int n0, int k0, int lane) {
    const half_t* p = BT + (size_t)(n0 + (lane & 15)) * kstride + k0 + ((lane >> 4) << 3);
    F16x16 f;
    f.u[0] = *(const uv4*)(p);
    f.u[1] = *(const uv4*)(p + 16);
    return f.h;
}

// D = tanh(A[16x32] @ B + bias), output f16 row-major 16x128 in LDS.
__device__ __forceinline__ void gemm_k32_tanh(const half_t* A, const half_t* BT,
                                              const float* __restrict__ bias,
                                              half_t* D, int lane) {
    v16h a = frag_a(A, 32, 0, lane);
    const int nl = lane & 15, rb = (lane >> 4) << 3;
#pragma unroll
    for (int t = 0; t < 8; ++t) {
        v16h b = frag_b(BT, 32, t * 16, 0, lane);
        v8f acc = {};
        acc = WMMA_F16(a, b, acc);
        const int col = t * 16 + nl;
        const float bv = bias[col];
#pragma unroll
        for (int r = 0; r < 8; ++r)
            D[(rb + r) * 128 + col] = (half_t)act_tanh(acc[r] + bv);
    }
}

// softmax-attention over nEnt entities for 16 rows.
// q: 16x128 f16 query ; emb: nEnt x 16 x 128 f16 ; att: 16x128 f16 out.
__device__ __forceinline__ void attention(const half_t* q, const half_t* emb, int nEnt,
                                          half_t* att, int lane) {
    const int row = lane >> 1;          // each row handled by a lane pair
    const int ch  = (lane & 1) * 64;    // column half
    float s[3];
    for (int e = 0; e < nEnt; ++e) {
        const half_t* er = emb + (e * 16 + row) * 128 + ch;
        const half_t* qr = q + row * 128 + ch;
        float p = 0.f;
#pragma unroll
        for (int c8 = 0; c8 < 8; ++c8) {
            v8h qa = *(const v8h*)(qr + c8 * 8);
            v8h ea = *(const v8h*)(er + c8 * 8);
#pragma unroll
            for (int j = 0; j < 8; ++j) p += (float)qa[j] * (float)ea[j];
        }
        p += __shfl_xor(p, 1, 32);                 // combine column halves
        s[e] = p * 0.08838834764831845f;           // 1/sqrt(128)
    }
    float m = s[0];
    for (int e = 1; e < nEnt; ++e) m = fmaxf(m, s[e]);
    float w[3], sum = 0.f;
    for (int e = 0; e < nEnt; ++e) {
        w[e] = __builtin_amdgcn_exp2f((s[e] - m) * 1.4426950408889634f);
        sum += w[e];
    }
    const float inv = 1.f / sum;
    half_t wh[3];
    for (int e = 0; e < nEnt; ++e) wh[e] = (half_t)(w[e] * inv);
    // weighted sum in packed f16 (v_pk_fma_f16)
#pragma unroll
    for (int c8 = 0; c8 < 8; ++c8) {
        v8h o = {};
        for (int e = 0; e < nEnt; ++e) {
            v8h ea = *(const v8h*)(emb + (e * 16 + row) * 128 + ch + c8 * 8);
            o += ea * wh[e];
        }
        *(v8h*)(att + row * 128 + ch + c8 * 8) = o;
    }
}

// ---------------- weight prepack: f32 -> f16, transposed, K-padded ----------------
__global__ __launch_bounds__(256) void prep_weights(
    const float* __restrict__ selfW, const float* __restrict__ enemyW,
    const float* __restrict__ allyW, const float* __restrict__ emat,
    const float* __restrict__ amat,  const float* __restrict__ enc1W,
    const float* __restrict__ enc2W, half_t* __restrict__ ws) {
    int i = blockIdx.x * blockDim.x + threadIdx.x;
    if (i >= WS_TOTAL) return;
    float v;
    if (i < WS_ENW)        { int j = i;            int n = j >> 5, k = j & 31;   v = (k < 17) ? selfW[k * 128 + n]  : 0.f; }
    else if (i < WS_ALW)   { int j = i - WS_ENW;   int n = j >> 5, k = j & 31;   v = (k < 5)  ? enemyW[k * 128 + n] : 0.f; }
    else if (i < WS_EMAT)  { int j = i - WS_ALW;   int n = j >> 5, k = j & 31;   v = (k < 5)  ? allyW[k * 128 + n]  : 0.f; }
    else if (i < WS_AMAT)  { int j = i - WS_EMAT;  int n = j >> 7, k = j & 127;  v = emat[k * 128 + n]; }
    else if (i < WS_ENC1)  { int j = i - WS_AMAT;  int n = j >> 7, k = j & 127;  v = amat[k * 128 + n]; }
    else if (i < WS_ENC2)  { int j = i - WS_ENC1;  int n = j / 384, k = j - n * 384; v = enc1W[k * 128 + n]; }
    else                   { int j = i - WS_ENC2;  int n = j >> 7, k = j & 127;  v = enc2W[k * 128 + n]; }
    ws[i] = (half_t)v;
}

// ---------------- main kernel: 4 waves/WG, 16 rows per wave ----------------
__global__ __launch_bounds__(128) void obs_encoder_main(
    const float* __restrict__ in,
    const float* __restrict__ self_b, const float* __restrict__ enemy_b,
    const float* __restrict__ ally_b, const float* __restrict__ enc1_b,
    const float* __restrict__ enc2_b,
    const half_t* __restrict__ W, float* __restrict__ out) {
    extern __shared__ char smem[];
    const int lane = threadIdx.x & 31;
    const int wave = threadIdx.x >> 5;

    half_t* wls  = (half_t*)(smem + wave * LDS_PER_WAVE);
    half_t* bufA = wls;                 // selfEmb  16x128      (2048 h)
    half_t* bufB = wls + 2048;          // enemyEmb 3x16x128    (6144 h)
    half_t* bufC = wls + 8192;          // allyEmb  2x16x128    (4096 h)
    half_t* bufD = wls + 12288;         // 6144 h, phase-aliased:
    half_t* stage = bufD;               //  ph1: selfA@0(512) enemyA@512(1536) allyA@2048(1024)
    half_t* qE    = bufD;               //  ph3: e-query 16x128
    half_t* qA    = bufD + 2048;        //  ph3: a-query 16x128
    half_t* attE  = bufD + 4096;        //  ph4: 16x128 (old q scratch dead)
    half_t* attA  = bufD;               //  ph4: 16x128 (qE dead)
    half_t* h2    = bufD + 2048;        //  ph5: 16x128 (qA dead)

    const long row0 = (long)blockIdx.x * (WAVES_PER_WG * ROWS_PER_WAVE) + wave * ROWS_PER_WAVE;
    const int nl = lane & 15, rb = (lane >> 4) << 3;

    // ---- phase 1: stage inputs as K-padded f16 A-tiles ----
    {
        unsigned* z = (unsigned*)stage;           // zero 3072 halves (padding)
        for (int i = lane; i < 1536; i += 32) z[i] = 0u;
        const float* inr = in + row0 * OBS_DIM;   // 16 contiguous rows
        for (int i = lane; i < 16 * OBS_DIM; i += 32) {
            int r = i / OBS_DIM, c = i - r * OBS_DIM;
            half_t h = (half_t)inr[i];
            if (c < 4)        stage[r * 32 + c] = h;                                  // move
            else if (c < 19)  { int e = (c - 4) / 5,  f = (c - 4) - e * 5;
                                stage[512 + (e * 16 + r) * 32 + f] = h; }             // enemies
            else if (c < 29)  { int a = (c - 19) / 5, f = (c - 19) - a * 5;
                                stage[2048 + (a * 16 + r) * 32 + f] = h; }            // allies
            else if (c == 29) stage[r * 32 + 4] = h;                                  // own
            else              stage[r * 32 + 5 + (c - 30)] = h;                       // last act/agent
        }
    }

    // ---- phase 2: entity embeddings (tanh) ----
    gemm_k32_tanh(stage,               W + WS_SELFW, self_b,  bufA, lane);
    for (int e = 0; e < 3; ++e)
        gemm_k32_tanh(stage + 512 + e * 512,  W + WS_ENW, enemy_b, bufB + e * 2048, lane);
    for (int a = 0; a < 2; ++a)
        gemm_k32_tanh(stage + 2048 + a * 512, W + WS_ALW, ally_b,  bufC + a * 2048, lane);

    // ---- phase 3: fused qE = selfEmb@enemy_mat, qA = selfEmb@ally_mat ----
    {
        v16h aS[4];
#pragma unroll
        for (int kk = 0; kk < 4; ++kk) aS[kk] = frag_a(bufA, 128, kk * 32, lane);
#pragma unroll
        for (int t = 0; t < 8; ++t) {
            v8f accE = {}, accA = {};
#pragma unroll
            for (int kk = 0; kk < 4; ++kk) {
                v16h bE = frag_b(W + WS_EMAT, 128, t * 16, kk * 32, lane);
                v16h bA = frag_b(W + WS_AMAT, 128, t * 16, kk * 32, lane);
                accE = WMMA_F16(aS[kk], bE, accE);
                accA = WMMA_F16(aS[kk], bA, accA);
            }
            const int col = t * 16 + nl;
#pragma unroll
            for (int r = 0; r < 8; ++r) {
                qE[(rb + r) * 128 + col] = (half_t)accE[r];
                qA[(rb + r) * 128 + col] = (half_t)accA[r];
            }
        }
    }

    // ---- phase 4: softmax attention ----
    attention(qE, bufB, 3, attE, lane);   // writes bufD+4096
    attention(qA, bufC, 2, attA, lane);   // writes bufD+0 (qE dead)

    // ---- phase 5: enc1 = tanh([self|attE|attA] @ enc1_W + b) ----
    {
        v16h aS[4], aE[4], aA[4];
#pragma unroll
        for (int kk = 0; kk < 4; ++kk) {
            aS[kk] = frag_a(bufA, 128, kk * 32, lane);
            aE[kk] = frag_a(attE, 128, kk * 32, lane);
            aA[kk] = frag_a(attA, 128, kk * 32, lane);
        }
#pragma unroll
        for (int t = 0; t < 8; ++t) {
            v8f acc = {};
#pragma unroll
            for (int kk = 0; kk < 4; ++kk)
                acc = WMMA_F16(aS[kk], frag_b(W + WS_ENC1, 384, t * 16, kk * 32, lane), acc);
#pragma unroll
            for (int kk = 0; kk < 4; ++kk)
                acc = WMMA_F16(aE[kk], frag_b(W + WS_ENC1, 384, t * 16, 128 + kk * 32, lane), acc);
#pragma unroll
            for (int kk = 0; kk < 4; ++kk)
                acc = WMMA_F16(aA[kk], frag_b(W + WS_ENC1, 384, t * 16, 256 + kk * 32, lane), acc);
            const int col = t * 16 + nl;
            const float bv = enc1_b[col];
#pragma unroll
            for (int r = 0; r < 8; ++r)
                h2[(rb + r) * 128 + col] = (half_t)act_tanh(acc[r] + bv);
        }
    }

    // ---- phase 6: out = tanh(h2 @ enc2_W + b), f32 to global ----
    {
        v16h aH[4];
#pragma unroll
        for (int kk = 0; kk < 4; ++kk) aH[kk] = frag_a(h2, 128, kk * 32, lane);
#pragma unroll
        for (int t = 0; t < 8; ++t) {
            v8f acc = {};
#pragma unroll
            for (int kk = 0; kk < 4; ++kk)
                acc = WMMA_F16(aH[kk], frag_b(W + WS_ENC2, 128, t * 16, kk * 32, lane), acc);
            const int col = t * 16 + nl;
            const float bv = enc2_b[col];
#pragma unroll
            for (int r = 0; r < 8; ++r)
                out[(size_t)(row0 + rb + r) * 128 + col] = act_tanh(acc[r] + bv);
        }
    }
}

// ---------------- launcher ----------------
extern "C" void kernel_launch(void* const* d_in, const int* in_sizes, int n_in,
                              void* d_out, int out_size, void* d_ws, size_t ws_size,
                              hipStream_t stream) {
    (void)in_sizes; (void)n_in; (void)out_size; (void)ws_size;
    const float* in      = (const float*)d_in[0];
    const float* self_W  = (const float*)d_in[1];
    const float* self_b  = (const float*)d_in[2];
    const float* enemy_W = (const float*)d_in[3];
    const float* enemy_b = (const float*)d_in[4];
    const float* ally_W  = (const float*)d_in[5];
    const float* ally_b  = (const float*)d_in[6];
    const float* emat    = (const float*)d_in[7];
    const float* amat    = (const float*)d_in[8];
    const float* enc1_W  = (const float*)d_in[9];
    const float* enc1_b  = (const float*)d_in[10];
    const float* enc2_W  = (const float*)d_in[11];
    const float* enc2_b  = (const float*)d_in[12];
    half_t* ws = (half_t*)d_ws;

    prep_weights<<<(WS_TOTAL + 255) / 256, 256, 0, stream>>>(
        self_W, enemy_W, ally_W, emat, amat, enc1_W, enc2_W, ws);

    (void)hipFuncSetAttribute((const void*)obs_encoder_main,
                              hipFuncAttributeMaxDynamicSharedMemorySize, LDS_TOTAL);
    const int rows_per_wg = WAVES_PER_WG * ROWS_PER_WAVE;  // 64
    obs_encoder_main<<<BATCH / rows_per_wg, WAVES_PER_WG * 32, LDS_TOTAL, stream>>>(
        in, self_b, enemy_b, ally_b, enc1_b, enc2_b, ws, (float*)d_out);
}
